// ExpertPancreasSwin3D_87316685128500
// MI455X (gfx1250) — matrix operand order repaired and checked by reading, not compiled, not run on hardware
//
#include <hip/hip_runtime.h>
#include <hip/hip_bf16.h>
#include <cstddef>

// ---------------------------------------------------------------------------
// Swin-3D forward for MI455X (gfx1250, wave32, WMMA).
// GEMMs run on v_wmma_f32_16x16x32_f16 with f16-resident activations and
// fragment-layout pre-packed f16 weights (zero-padded K). The GEMM inner loop
// is software-pipelined b128 loads + WMMA with no per-chunk branches.
// f32 residual stream, f32 accumulation.
// ---------------------------------------------------------------------------

typedef __attribute__((ext_vector_type(16))) _Float16 v16h;
typedef __attribute__((ext_vector_type(8)))  _Float16 v8h;
typedef __attribute__((ext_vector_type(8)))  float    v8f;

// B fragment: 16 contiguous halves (2 x b128)
static __device__ inline v16h swin_ld_frag(const _Float16* p)
{
    v8h x0 = *(const v8h*)p;
    v8h x1 = *(const v8h*)(p + 8);
    v16h r;
#pragma unroll
    for (int e = 0; e < 8; ++e) { r[e] = x0[e]; r[8 + e] = x1[e]; }
    return r;
}

// A fragment: halves at +0..7 and +16..23 relative to (row base + half*8)
static __device__ inline v16h swin_ld_fragA(const _Float16* p)
{
    v8h x0 = *(const v8h*)p;
    v8h x1 = *(const v8h*)(p + 16);
    v16h r;
#pragma unroll
    for (int e = 0; e < 8; ++e) { r[e] = x0[e]; r[8 + e] = x1[e]; }
    return r;
}

// ---------------------------------------------------------------------------
// Weight pre-pack: W[K,N] f32 (row-major) -> fragment-major f16:
//   P[((nt*KC + kc)*32 + lane)*16 + e] = W[kc*32 + (e>>3)*16 + (lane>>4)*8 + (e&7)][nt*16 + (lane&15)]
// with K zero-padded to KC*32. A wave's B fragment = 2 x b128 per chunk.
// ---------------------------------------------------------------------------
__global__ void swin_packw_kernel(const float* __restrict__ W,
                                  _Float16* __restrict__ P,
                                  int K, int N, int KC)
{
    int idx = blockIdx.x * blockDim.x + threadIdx.x;
    int total = (N >> 4) * KC * 512;
    if (idx >= total) return;
    int e    = idx & 15;
    int l    = (idx >> 4) & 31;
    int rest = idx >> 9;
    int kc   = rest % KC;
    int nt   = rest / KC;
    int k    = kc * 32 + (e >> 3) * 16 + (l >> 4) * 8 + (e & 7);
    int col  = nt * 16 + (l & 15);
    P[idx] = (k < K) ? (_Float16)W[(size_t)k * N + col] : (_Float16)0.f;
}

// ---------------------------------------------------------------------------
// Generic GEMM: out[M,N] = act(A[M,K] @ W[K,N] + bias) (+ res)
// A f16, Wp pre-packed f16 (K padded), bias/res f32, OutT float or _Float16.
// TWO: wave computes two adjacent N-tiles sharing one A fragment.
// One-chunk software pipeline: next fragments are in flight during the WMMAs.
// act: 0 = none, 1 = exact GELU.
// ---------------------------------------------------------------------------
template <typename OutT, bool TWO>
__global__ void swin_gemm_kernel(const _Float16* __restrict__ A,
                                 const _Float16* __restrict__ Wp,
                                 const float* __restrict__ bias,
                                 const float* __restrict__ res,
                                 OutT* __restrict__ out,
                                 int M, int N, int K, int act, int ntOff)
{
    const int lane = threadIdx.x & 31;
    const int half = lane >> 4;
    const int lm   = lane & 15;
    const int mrow = blockIdx.x * 16;
    const int KC   = (K + 31) >> 5;
    const int nt0  = TWO ? (int)blockIdx.y * 2 : ntOff;
    const int nc0  = nt0 * 16;
    const int nc1  = nc0 + 16;

    v8f c0 = {}, c1 = {};
    if (bias) {
        float b0 = bias[nc0 + lm];
#pragma unroll
        for (int r = 0; r < 8; ++r) c0[r] = b0;
        if constexpr (TWO) {
            float b1 = bias[nc1 + lm];
#pragma unroll
            for (int r = 0; r < 8; ++r) c1[r] = b1;
        }
    }

    const _Float16* arow = A + (size_t)(mrow + lm) * K + half * 8;
    const _Float16* bp0  = Wp + ((size_t)nt0 * KC * 32 + lane) * 16;
    const _Float16* bp1  = bp0 + (size_t)KC * 512;   // adjacent N-tile

    const int kcfull = K >> 5;
    v16h a_c{}, b_c{}, d_c{};
    if (kcfull > 0) {
        a_c = swin_ld_fragA(arow);
        b_c = swin_ld_frag(bp0);
        if constexpr (TWO) d_c = swin_ld_frag(bp1);
    }
    for (int i = 0; i < kcfull - 1; ++i) {
        // prefetch chunk i+1 while WMMAs consume chunk i
        v16h a_n = swin_ld_fragA(arow + (i + 1) * 32);
        v16h b_n = swin_ld_frag(bp0 + (size_t)(i + 1) * 512);
        v16h d_n{};
        if constexpr (TWO) d_n = swin_ld_frag(bp1 + (size_t)(i + 1) * 512);
        c0 = __builtin_amdgcn_wmma_f32_16x16x32_f16(false, a_c, false, b_c, (short)0,
                                                    c0, false, false);
        if constexpr (TWO)
            c1 = __builtin_amdgcn_wmma_f32_16x16x32_f16(false, a_c, false, d_c,
                                                        (short)0, c1, false, false);
        a_c = a_n; b_c = b_n;
        if constexpr (TWO) d_c = d_n;
    }
    if (kcfull > 0) {
        c0 = __builtin_amdgcn_wmma_f32_16x16x32_f16(false, a_c, false, b_c, (short)0,
                                                    c0, false, false);
        if constexpr (TWO)
            c1 = __builtin_amdgcn_wmma_f32_16x16x32_f16(false, a_c, false, d_c,
                                                        (short)0, c1, false, false);
    }

    if (K & 16) {                            // uniform 16-wide A tail; Wp is padded
        v8h a0 = *(const v8h*)(arow + kcfull * 32);
        v16h a;
#pragma unroll
        for (int e = 0; e < 8; ++e) { a[e] = a0[e]; a[8 + e] = (_Float16)0.f; }
        v16h b = swin_ld_frag(bp0 + (size_t)kcfull * 512);
        c0 = __builtin_amdgcn_wmma_f32_16x16x32_f16(false, a, false, b, (short)0, c0,
                                                    false, false);
        if constexpr (TWO) {
            v16h b2 = swin_ld_frag(bp1 + (size_t)kcfull * 512);
            c1 = __builtin_amdgcn_wmma_f32_16x16x32_f16(false, a, false, b2, (short)0,
                                                        c1, false, false);
        }
    }

#pragma unroll
    for (int r = 0; r < 8; ++r) {
        size_t idx = (size_t)(mrow + r + 8 * half) * N + nc0 + lm;
        float v = c0[r];
        if (act == 1) v = 0.5f * v * (1.f + erff(v * 0.70710678118654752f));
        if (res) v += res[idx];
        out[idx] = (OutT)v;
    }
    if constexpr (TWO) {
#pragma unroll
        for (int r = 0; r < 8; ++r) {
            size_t idx = (size_t)(mrow + r + 8 * half) * N + nc1 + lm;
            float v = c1[r];
            if (act == 1) v = 0.5f * v * (1.f + erff(v * 0.70710678118654752f));
            if (res) v += res[idx];
            out[idx] = (OutT)v;
        }
    }
}

// ---------------------------------------------------------------------------
// Window attention, templated on window shape (all power-of-2 dims).
// One workgroup per (window, head); one wave per 16-row score tile.
// qkv/out are f16; softmax in f32 registers; hd == 16, scale 0.25 (exact f16).
// ---------------------------------------------------------------------------
template <int LW0, int LW1, int LW2>
__global__ void swin_attn_kernel(const _Float16* __restrict__ qkv,
                                 _Float16* __restrict__ out,
                                 const float* __restrict__ bias_tbl,
                                 int Hn, int C, int nWin,
                                 int nH, int nW,
                                 int T, int Hh, int Wd,
                                 int s0, int s1, int s2, int masked)
{
    constexpr int NT    = LW0 * LW1 * LW2;   // 64 or 16
    constexpr int NTILE = NT / 16;
    constexpr int PS    = NT + 8;
    __shared__ _Float16 p_lds[NTILE * 16 * PS];

    const int lane = threadIdx.x & 31;
    const int wv   = threadIdx.x >> 5;
    const int half = lane >> 4;
    const int lm   = lane & 15;
    const int widx = blockIdx.x / Hn;        // uniform
    const int h    = blockIdx.x % Hn;        // uniform
    const int threeC = 3 * C;

    const _Float16* qp = qkv + (size_t)widx * NT * threeC + h * 16;
    const _Float16* kp = qp + C;
    const _Float16* vp = qp + 2 * C;

    // ---- q A-fragment (K padded 16 -> 32): one b128, upper half zero ----
    v16h aq;
    {
        v8h q8 = *(const v8h*)(qp + (size_t)(wv * 16 + lm) * threeC + half * 8);
#pragma unroll
        for (int e = 0; e < 8; ++e) {
            aq[e]     = q8[e] * (_Float16)0.25f;
            aq[8 + e] = (_Float16)0.f;
        }
    }

    // ---- S = q @ k^T ----
    v8f s[NTILE];
#pragma unroll
    for (int ct = 0; ct < NTILE; ++ct) {
        const _Float16* krow = kp + (size_t)(ct * 16 + lm) * threeC;
        v8h k0 = *(const v8h*)(krow);
        v8h k1 = *(const v8h*)(krow + 8);
        v16h bk;
#pragma unroll
        for (int e = 0; e < 8; ++e) {
            bk[e]     = half ? k1[e] : k0[e];
            bk[8 + e] = (_Float16)0.f;
        }
        v8f z = {};
        s[ct] = __builtin_amdgcn_wmma_f32_16x16x32_f16(false, aq, false, bk, (short)0,
                                                       z, false, false);
    }

    // ---- relative-position bias + shifted-window mask ----
    const int win = widx % nWin;             // uniform
    const int wt  = win / (nH * nW);
    const int wh  = (win / nW) % nH;
    const int wwi = win % nW;

    auto region = [](int cc, int size, int wdim, int sdim) -> int {
        if (sdim == 0) return 0;
        return (cc < size - wdim) ? 0 : ((cc < size - sdim) ? 1 : 2);
    };
    auto label = [&](int a0, int a1, int a2) -> int {
        int c0 = wt * LW0 + a0 + s0; if (c0 >= T)  c0 -= T;
        int c1 = wh * LW1 + a1 + s1; if (c1 >= Hh) c1 -= Hh;
        int c2 = wwi * LW2 + a2 + s2; if (c2 >= Wd) c2 -= Wd;
        int m1 = s1 ? 3 : 1, m2 = s2 ? 3 : 1;
        return (region(c0, T, LW0, s0) * m1 + region(c1, Hh, LW1, s1)) * m2 +
               region(c2, Wd, LW2, s2);
    };

    int jb0[NTILE], jb1[NTILE], jb2[NTILE], jl[NTILE];
#pragma unroll
    for (int ct = 0; ct < NTILE; ++ct) {
        int j = ct * 16 + lm;
        jb2[ct] = j % LW2;                    // constant pow2 -> mask
        jb1[ct] = (j / LW2) % LW1;
        jb0[ct] = j / (LW1 * LW2);
        jl[ct]  = masked ? label(jb0[ct], jb1[ct], jb2[ct]) : 0;
    }

#pragma unroll
    for (int r = 0; r < 8; ++r) {
        int i  = wv * 16 + r + 8 * half;
        int a2 = i % LW2, a1 = (i / LW2) % LW1, a0 = i / (LW1 * LW2);
        int li = masked ? label(a0, a1, a2) : 0;
#pragma unroll
        for (int ct = 0; ct < NTILE; ++ct) {
            int rel = ((a0 - jb0[ct] + LW0 - 1) * (2 * LW1 - 1) +
                       (a1 - jb1[ct] + LW1 - 1)) * (2 * LW2 - 1) +
                      (a2 - jb2[ct] + LW2 - 1);
            float bb = bias_tbl[(size_t)rel * Hn + h];
            if (masked && li != jl[ct]) bb -= 100.f;
            s[ct][r] += bb;
        }
    }

    // ---- row softmax (row lives on a fixed 16-lane half-group) ----
#pragma unroll
    for (int r = 0; r < 8; ++r) {
        float mx = -1e30f;
#pragma unroll
        for (int ct = 0; ct < NTILE; ++ct) mx = fmaxf(mx, s[ct][r]);
        mx = fmaxf(mx, __shfl_xor(mx, 1));
        mx = fmaxf(mx, __shfl_xor(mx, 2));
        mx = fmaxf(mx, __shfl_xor(mx, 4));
        mx = fmaxf(mx, __shfl_xor(mx, 8));
        float sum = 0.f;
#pragma unroll
        for (int ct = 0; ct < NTILE; ++ct) {
            float e = __expf(s[ct][r] - mx);
            s[ct][r] = e;
            sum += e;
        }
        sum += __shfl_xor(sum, 1);
        sum += __shfl_xor(sum, 2);
        sum += __shfl_xor(sum, 4);
        sum += __shfl_xor(sum, 8);
        float inv = 1.f / sum;
#pragma unroll
        for (int ct = 0; ct < NTILE; ++ct)
            p_lds[(wv * 16 + r + 8 * half) * PS + ct * 16 + lm] =
                (_Float16)(s[ct][r] * inv);
    }
    __syncthreads();

    // ---- out = P @ v  (K = NT, chained 16x16x32 WMMAs) ----
    v8f o = {};
    constexpr int KCH = (NT + 31) / 32;
#pragma unroll
    for (int cch = 0; cch < KCH; ++cch) {
        v16h ap;
#pragma unroll
        for (int v = 0; v < 8; ++v) {
            if (NT == 16 && v >= 4) {        // compile-time: K pad
                ap[2*v] = (_Float16)0.f; ap[2*v+1] = (_Float16)0.f;
            } else {
                int k = (NT == 16) ? (half * 8 + (v & 3) * 2)
                                   : (cch * 32 + (v >> 2) * 16 + half * 8 + (v & 3) * 2);
                ap[2*v]   = p_lds[(wv * 16 + lm) * PS + k];
                ap[2*v+1] = p_lds[(wv * 16 + lm) * PS + k + 1];
            }
        }
        v16h bv;
#pragma unroll
        for (int v = 0; v < 8; ++v) {
            if (NT == 16) {                  // elements 8-15 carry K pad -> zero
                int krow = half * 8 + 2 * (v & 3);
                _Float16 v0 = vp[(size_t)krow * threeC + lm];
                _Float16 v1 = vp[(size_t)(krow + 1) * threeC + lm];
                bv[2*v]   = (v >= 4) ? (_Float16)0.f : v0;
                bv[2*v+1] = (v >= 4) ? (_Float16)0.f : v1;
            } else {                         // krow <= 62 < 64: always valid
                int krow = cch * 32 + (v >> 2) * 16 + half * 8 + 2 * (v & 3);
                bv[2*v]   = vp[(size_t)krow * threeC + lm];
                bv[2*v+1] = vp[(size_t)(krow + 1) * threeC + lm];
            }
        }
        o = __builtin_amdgcn_wmma_f32_16x16x32_f16(false, ap, false, bv, (short)0, o,
                                                   false, false);
    }

#pragma unroll
    for (int r = 0; r < 8; ++r)
        out[(size_t)(widx * NT + wv * 16 + r + 8 * half) * C + h * 16 + lm] =
            (_Float16)o[r];
}

// ---------------------------------------------------------------------------
// LayerNorm over last dim C: one wave per row. OutT = float or _Float16.
// ---------------------------------------------------------------------------
template <typename OutT>
__global__ void swin_ln_kernel(const float* __restrict__ x,
                               const float* __restrict__ g,
                               const float* __restrict__ b,
                               OutT* __restrict__ y, int C)
{
    const size_t row = blockIdx.x;
    const int lane = threadIdx.x;
    const float* xr = x + row * C;
    float s = 0.f, s2 = 0.f;
    for (int c = lane; c < C; c += 32) { float v = xr[c]; s += v; s2 += v * v; }
    for (int off = 16; off > 0; off >>= 1) {
        s  += __shfl_xor(s, off);
        s2 += __shfl_xor(s2, off);
    }
    float mean = s / C;
    float var  = s2 / C - mean * mean;
    float inv  = rsqrtf(var + 1e-5f);
    for (int c = lane; c < C; c += 32)
        y[row * C + c] = (OutT)((xr[c] - mean) * inv * g[c] + b[c]);
}

// ---------------------------------------------------------------------------
// Window partition with cyclic shift (f16, copied as packed u32).
// One block per window-ordered token row; decomposition is block-uniform.
// ---------------------------------------------------------------------------
__global__ void swin_winpart_kernel(const _Float16* __restrict__ x,
                                    _Float16* __restrict__ xw,
                                    int T, int Hh, int Wd, int C,
                                    int w0, int w1, int w2,
                                    int s0, int s1, int s2,
                                    int n, int nWin)
{
    const int wing = blockIdx.x;             // [B*nWin*n)
    const int loc  = wing % n;
    const int wq   = wing / n;
    const int win  = wq % nWin;
    const int b    = wq / nWin;
    const int nH = Hh / w1, nW = Wd / w2;
    const int a2 = loc % w2, a1 = (loc / w2) % w1, a0 = loc / (w1 * w2);
    const int wt = win / (nH * nW), wh = (win / nW) % nH, ww = win % nW;
    int t  = wt * w0 + a0 + s0; if (t >= T)  t -= T;
    int hh = wh * w1 + a1 + s1; if (hh >= Hh) hh -= Hh;
    int wd = ww * w2 + a2 + s2; if (wd >= Wd) wd -= Wd;
    const uint32_t* src = (const uint32_t*)
        (x + ((((size_t)b * T + t) * Hh + hh) * Wd + wd) * C);
    uint32_t* dst = (uint32_t*)(xw + (size_t)wing * C);
    const int C2 = C >> 1;
    for (int c = threadIdx.x; c < C2; c += blockDim.x) dst[c] = src[c];
}

// ---------------------------------------------------------------------------
// Window un-partition (inverse shift) + residual add (f32): out = res + proj
// ---------------------------------------------------------------------------
__global__ void swin_winunpart_kernel(const float* __restrict__ proj,
                                      const float* __restrict__ resx,
                                      float* __restrict__ out,
                                      int T, int Hh, int Wd, int C,
                                      int w0, int w1, int w2,
                                      int s0, int s1, int s2)
{
    int tok = blockIdx.x;                    // [B*T*Hh*Wd)
    const int wd = tok % Wd; tok /= Wd;
    const int hh = tok % Hh; tok /= Hh;
    const int t  = tok % T;
    const int b  = tok / T;
    int rt = t - s0;  if (rt < 0) rt += T;
    int rh = hh - s1; if (rh < 0) rh += Hh;
    int rw = wd - s2; if (rw < 0) rw += Wd;
    const int nH = Hh / w1, nW = Wd / w2;
    const int nWin = (T / w0) * nH * nW;
    const int n = w0 * w1 * w2;
    const int win = (rt / w0 * nH + rh / w1) * nW + rw / w2;
    const int loc = ((rt % w0) * w1 + rh % w1) * w2 + rw % w2;
    const float* src = proj + (((size_t)b * nWin + win) * n + loc) * C;
    const size_t drow = (size_t)blockIdx.x * C;
    for (int c = threadIdx.x; c < C; c += blockDim.x)
        out[drow + c] = resx[drow + c] + src[c];
}

// ---------------------------------------------------------------------------
// Patch merging concat: x[B,T,H,W,C] -> out[B,T,H/2,W/2,4C] (f32).
// ---------------------------------------------------------------------------
__global__ void swin_merge_kernel(const float* __restrict__ x,
                                  float* __restrict__ out,
                                  int T, int Hh, int Wd, int C)
{
    const int H2 = Hh / 2, W2 = Wd / 2;
    int row = blockIdx.x;                    // [B*T*H2*W2)
    const int w2i = row % W2; row /= W2;
    const int h2i = row % H2; row /= H2;
    const int t = row % T;
    const int b = row / T;
    float* dst = out + (size_t)blockIdx.x * 4 * C;
#pragma unroll
    for (int q = 0; q < 4; ++q) {
        const int hh = 2 * h2i + (q & 1);
        const int wd = 2 * w2i + (q >> 1);
        const float* src = x + ((((size_t)b * T + t) * Hh + hh) * Wd + wd) * C;
        for (int c = threadIdx.x; c < C; c += blockDim.x)
            dst[q * C + c] = src[c];
    }
}

// ---------------------------------------------------------------------------
// Patch embed: conv 4x4x4 stride 4, 1 -> 48 channels. One thread per (tok, co).
// ---------------------------------------------------------------------------
__global__ void swin_patch_kernel(const float* __restrict__ x,
                                  const float* __restrict__ pw,
                                  const float* __restrict__ pb,
                                  float* __restrict__ out)
{
    int id = blockIdx.x * blockDim.x + threadIdx.x;
    const int total = 64 * 4096 * 48;
    if (id >= total) return;
    int co = id % 48;
    int tok = id / 48;
    int b = tok >> 12;
    int t = (tok >> 8) & 15;
    int h = (tok >> 4) & 15;
    int w = tok & 15;
    const float* xb = x + ((((size_t)b * 64 + t * 4) * 64 + h * 4) * 64 + w * 4);
    const float* wc = pw + co * 64;
    float acc = pb[co];
#pragma unroll
    for (int i = 0; i < 4; ++i)
#pragma unroll
        for (int j = 0; j < 4; ++j)
#pragma unroll
            for (int k = 0; k < 4; ++k)
                acc += xb[((size_t)i * 64 + j) * 64 + k] * wc[(i * 4 + j) * 4 + k];
    out[(size_t)tok * 48 + co] = acc;
}

// ---------------------------------------------------------------------------
// Final pooled mean over 64 tokens per batch, then 384 -> 2 head.
// ---------------------------------------------------------------------------
__global__ void swin_pool_kernel(const float* __restrict__ x, float* __restrict__ out)
{
    int id = blockIdx.x * blockDim.x + threadIdx.x;
    if (id >= 64 * 384) return;
    int c = id % 384, b = id / 384;
    float s = 0.f;
    for (int t = 0; t < 64; ++t) s += x[((size_t)b * 64 + t) * 384 + c];
    out[id] = s * (1.f / 64.f);
}

__global__ void swin_head_kernel(const float* __restrict__ pooled,
                                 const float* __restrict__ hw,
                                 const float* __restrict__ hb,
                                 float* __restrict__ out)
{
    int id = blockIdx.x * blockDim.x + threadIdx.x;
    if (id >= 128) return;
    int o = id & 1, b = id >> 1;
    float s = hb[o];
    for (int c = 0; c < 384; ++c) s += pooled[b * 384 + c] * hw[c * 2 + o];
    out[id] = s;
}

// ---------------------------------------------------------------------------
// Host helpers
// ---------------------------------------------------------------------------
template <typename OutT>
static void launch_gemm(const _Float16* A, const _Float16* Wp, const float* bias,
                        const float* res, OutT* out, int M, int N, int K, int act,
                        hipStream_t stream)
{
    const int ntiles = N >> 4;
    if (ntiles >= 2)
        swin_gemm_kernel<OutT, true><<<dim3(M / 16, ntiles / 2), 32, 0, stream>>>(
            A, Wp, bias, res, out, M, N, K, act, 0);
    if (ntiles & 1)
        swin_gemm_kernel<OutT, false><<<dim3(M / 16, 1), 32, 0, stream>>>(
            A, Wp, bias, res, out, M, N, K, act, ntiles - 1);
}

// ---------------------------------------------------------------------------
// Host orchestration
// ---------------------------------------------------------------------------
extern "C" void kernel_launch(void* const* d_in, const int* in_sizes, int n_in,
                              void* d_out, int out_size, void* d_ws, size_t ws_size,
                              hipStream_t stream)
{
    (void)in_sizes; (void)n_in; (void)out_size; (void)ws_size;

    const float* x_in = (const float*)d_in[0];
    int pi = 1;
    auto NXp = [&]() { return (const float*)d_in[pi++]; };

    const float* patch_w  = NXp();
    const float* patch_b  = NXp();
    const float* patch_ng = NXp();
    const float* patch_nb = NXp();

    struct BlockP {
        const float *n1g, *n1b, *qkv_w, *qkv_b, *bias, *proj_w, *proj_b,
                    *n2g, *n2b, *m1w, *m1b, *m2w, *m2b;
    };
    BlockP bp[12];
    const float *mg_ng[3], *mg_nb[3], *mg_w[3];
    const int DEP[4] = {2, 2, 6, 2};
    {
        int bi = 0;
        for (int i = 0; i < 4; ++i) {
            for (int j = 0; j < DEP[i]; ++j) {
                BlockP& bl = bp[bi++];
                bl.n1g = NXp(); bl.n1b = NXp();
                bl.qkv_w = NXp(); bl.qkv_b = NXp();
                bl.bias = NXp();
                bl.proj_w = NXp(); bl.proj_b = NXp();
                bl.n2g = NXp(); bl.n2b = NXp();
                bl.m1w = NXp(); bl.m1b = NXp();
                bl.m2w = NXp(); bl.m2b = NXp();
            }
            if (i < 3) { mg_ng[i] = NXp(); mg_nb[i] = NXp(); mg_w[i] = NXp(); }
        }
    }
    const float* norm_g = NXp();
    const float* norm_b = NXp();
    const float* head_w = NXp();
    const float* head_b = NXp();

    // Workspace layout:
    //   o_x    f32 residual stream      12.6M f32
    //   o_pf   f32 proj/merge/final     12.6M f32
    //   o_lnh  f16 LN output            12.6M h
    //   o_xwh  f16 window buffer        12.6M h
    //   o_qkvh f16 qkv                  38.0M h
    //   o_hh   f16 MLP hidden           50.4M h
    //   wt_a   f16 packed-weight arena   2.0M h (reused per layer)
    float*    o_x    = (float*)d_ws;
    float*    o_pf   = o_x + 12600000;
    _Float16* o_lnh  = (_Float16*)(o_pf + 12600000);
    _Float16* o_xwh  = o_lnh + 12600000;
    _Float16* o_qkvh = o_xwh + 12600000;
    _Float16* o_hh   = o_qkvh + 38000000;
    _Float16* wt_a   = o_hh + 50400000;

    // pre-pack W[K,N] into fragment-major padded f16; returns size consumed
    auto PACK = [&](const float* srcW, int K, int N, _Float16* dst) -> int {
        int KC = (K + 31) >> 5;
        int tot = (N >> 4) * KC * 512;
        swin_packw_kernel<<<(tot + 255) / 256, 256, 0, stream>>>(srcW, dst, K, N, KC);
        return tot;
    };

    // ---- patch embed + LN (f32 residual stream) ----
    {
        int tot = 64 * 4096 * 48;
        swin_patch_kernel<<<(tot + 255) / 256, 256, 0, stream>>>(x_in, patch_w,
                                                                 patch_b, o_pf);
        swin_ln_kernel<float><<<64 * 4096, 32, 0, stream>>>(o_pf, patch_ng,
                                                            patch_nb, o_x, 48);
    }

    // ---- stages ----
    int T = 16, Hh = 16, Wd = 16, C = 48;
    const int HDS[4] = {3, 6, 12, 24};
    const int W0[4] = {4, 4, 4, 4}, W1[4] = {4, 4, 4, 2}, W2[4] = {4, 4, 4, 2};
    const int S0[4] = {2, 2, 2, 2}, S1[4] = {2, 2, 0, 0}, S2[4] = {2, 2, 0, 0};

    int bi = 0;
    for (int i = 0; i < 4; ++i) {
        const int w0 = W0[i], w1 = W1[i], w2 = W2[i];
        const int n = w0 * w1 * w2;
        const int nH = Hh / w1, nW = Wd / w2;
        const int nWin = (T / w0) * nH * nW;
        const int M = 64 * T * Hh * Wd;
        const int Hn = HDS[i];

        for (int j = 0; j < DEP[i]; ++j) {
            const BlockP& P = bp[bi++];
            const int sh = (j & 1);
            const int s0 = sh ? S0[i] : 0, s1 = sh ? S1[i] : 0, s2 = sh ? S2[i] : 0;
            const int masked = (s0 | s1 | s2) ? 1 : 0;

            // pack this block's weights into the arena (stream-ordered reuse)
            _Float16* wp = wt_a;
            const _Float16* qkvWh  = wp; wp += PACK(P.qkv_w,  C,     3 * C, wp);
            const _Float16* projWh = wp; wp += PACK(P.proj_w, C,     C,     wp);
            const _Float16* m1Wh   = wp; wp += PACK(P.m1w,    C,     4 * C, wp);
            const _Float16* m2Wh   = wp; wp += PACK(P.m2w,    4 * C, C,     wp);

            swin_ln_kernel<_Float16><<<M, 32, 0, stream>>>(o_x, P.n1g, P.n1b,
                                                           o_lnh, C);
            swin_winpart_kernel<<<M, 64, 0, stream>>>(o_lnh, o_xwh, T, Hh, Wd, C,
                                                      w0, w1, w2, s0, s1, s2,
                                                      n, nWin);

            launch_gemm<_Float16>(o_xwh, qkvWh, P.qkv_b, nullptr, o_qkvh,
                                  M, 3 * C, C, 0, stream);

            const int blocks = (M / n) * Hn;
            if (n == 64)
                swin_attn_kernel<4, 4, 4><<<blocks, 128, 0, stream>>>(
                    o_qkvh, o_xwh, P.bias, Hn, C, nWin, nH, nW,
                    T, Hh, Wd, s0, s1, s2, masked);
            else
                swin_attn_kernel<4, 2, 2><<<blocks, 32, 0, stream>>>(
                    o_qkvh, o_xwh, P.bias, Hn, C, nWin, nH, nW,
                    T, Hh, Wd, s0, s1, s2, masked);

            launch_gemm<float>(o_xwh, projWh, P.proj_b, nullptr, o_pf,
                               M, C, C, 0, stream);
            swin_winunpart_kernel<<<M, 64, 0, stream>>>(o_pf, o_x, o_x,
                                                        T, Hh, Wd, C,
                                                        w0, w1, w2, s0, s1, s2);

            swin_ln_kernel<_Float16><<<M, 32, 0, stream>>>(o_x, P.n2g, P.n2b,
                                                           o_lnh, C);
            launch_gemm<_Float16>(o_lnh, m1Wh, P.m1b, nullptr, o_hh,
                                  M, 4 * C, C, 1, stream);
            launch_gemm<float>(o_hh, m2Wh, P.m2b, o_x, o_x,
                               M, C, 4 * C, 0, stream);
        }

        if (i < 3) {
            const int Mn = 64 * T * (Hh / 2) * (Wd / 2);
            const _Float16* mgWh = wt_a;
            PACK(mg_w[i], 4 * C, 2 * C, wt_a);
            swin_merge_kernel<<<Mn, 64, 0, stream>>>(o_x, o_pf, T, Hh, Wd, C);
            swin_ln_kernel<_Float16><<<Mn, 32, 0, stream>>>(o_pf, mg_ng[i],
                                                            mg_nb[i], o_lnh, 4 * C);
            launch_gemm<float>(o_lnh, mgWh, nullptr, nullptr, o_x,
                               Mn, 2 * C, 4 * C, 0, stream);
            Hh /= 2; Wd /= 2; C *= 2;
        }
    }

    // ---- final LN, pool, head ----
    swin_ln_kernel<float><<<64 * 64, 32, 0, stream>>>(o_x, norm_g, norm_b,
                                                      o_pf, 384);
    swin_pool_kernel<<<(64 * 384 + 255) / 256, 256, 0, stream>>>(o_pf,
                                                                 (float*)o_xwh);
    swin_head_kernel<<<1, 128, 0, stream>>>((float*)o_xwh, head_w, head_b,
                                            (float*)d_out);
}
